// MultiHeadAttention_39582418599997
// MI455X (gfx1250) — matrix operand order repaired and checked by reading, not compiled, not run on hardware
//
#include <hip/hip_runtime.h>
#include <hip/hip_bf16.h>
#include <cstdint>

#define D_MODEL 512
#define S_LEN   2048
#define NB      4
#define NH      8
#define DKH     64
#define MROWS   (NB * S_LEN)   // 8192

typedef __attribute__((ext_vector_type(16))) __bf16 bf16x16;
typedef __attribute__((ext_vector_type(8)))  float  f32x8;

union FragB { uint32_t u[8]; bf16x16 v; };

// Full RNE pack (used only in bandwidth-bound pack kernels where VALU is free).
__device__ inline uint32_t pack_bf16_rne(float lo, float hi) {
    uint32_t a = __builtin_bit_cast(uint32_t, lo);
    uint32_t b = __builtin_bit_cast(uint32_t, hi);
    a += 0x7fffu + ((a >> 16) & 1u);
    b += 0x7fffu + ((b >> 16) & 1u);
    return (a >> 16) | (b & 0xffff0000u);
}
// Cheap 2-op round-half-up convert for hot-loop scalar stores.
__device__ inline uint16_t bf16u(float f) {
    uint32_t x = __builtin_bit_cast(uint32_t, f);
    return (uint16_t)((x + 0x8000u) >> 16);
}
__device__ inline f32x8 wmma_bf16(const FragB& a, const FragB& b, f32x8 c) {
    return __builtin_amdgcn_wmma_f32_16x16x32_bf16(
        false, a.v, false, b.v, (short)0, c, false, false);
}
__device__ inline float rmax16(float v) {
    v = fmaxf(v, __shfl_xor(v, 1, 32));
    v = fmaxf(v, __shfl_xor(v, 2, 32));
    v = fmaxf(v, __shfl_xor(v, 4, 32));
    v = fmaxf(v, __shfl_xor(v, 8, 32));
    return v;
}
__device__ inline float rsum16(float v) {
    v += __shfl_xor(v, 1, 32);
    v += __shfl_xor(v, 2, 32);
    v += __shfl_xor(v, 4, 32);
    v += __shfl_xor(v, 8, 32);
    return v;
}

// Async-copy a contiguous 4KB tile (32 keys x 64 bf16) global -> LDS.
// 8 x GLOBAL_LOAD_ASYNC_TO_LDS_B128: each moves 16B/lane x 32 lanes = 512B.
// Tracked by ASYNCcnt; caller gates with s_wait_asynccnt.
__device__ inline void async_copy_4k(const uint16_t* gsrc, uint16_t* ldst, int lane) {
    uint32_t lds0 = (uint32_t)(size_t)ldst;          // generic->LDS offset (addr[31:0])
    const char* g0 = (const char*)gsrc;
    #pragma unroll
    for (int t = 0; t < 8; ++t) {
        uint32_t loff  = lds0 + (uint32_t)(t * 512 + lane * 16);
        uint64_t gaddr = (uint64_t)(size_t)(g0 + t * 512 + lane * 16);
        asm volatile("global_load_async_to_lds_b128 %0, %1, off"
                     :: "v"(loff), "v"(gaddr) : "memory");
    }
}
__device__ inline void wait_async_le16() {
    asm volatile("s_wait_asynccnt 16" ::: "memory");
}
__device__ inline void wait_async_0() {
    asm volatile("s_wait_asynccnt 0" ::: "memory");
}

// ---------------------------------------------------------------------------
// Repack fp32 weight [K=512, N=512] -> bf16 with K-pairs interleaved:
// Wp_u32[(k/2)*N + n] = { bf16(W[k][n]) , bf16(W[k+1][n]) }  (lo = even k)
// ---------------------------------------------------------------------------
__global__ void mha_pack_w(const float* __restrict__ W, uint32_t* __restrict__ Wp) {
    int idx = blockIdx.x * blockDim.x + threadIdx.x;     // over (K/2)*N
    if (idx >= (D_MODEL / 2) * D_MODEL) return;
    int p = idx / D_MODEL, n = idx % D_MODEL;
    float lo = W[(2 * p) * D_MODEL + n];
    float hi = W[(2 * p + 1) * D_MODEL + n];
    Wp[p * D_MODEL + n] = pack_bf16_rne(lo, hi);
}

// ---------------------------------------------------------------------------
// Pack fp32 activations [8192,512] row-major -> bf16 pair-packed dwords
// [8192, 256].  Pairs along K are already contiguous in memory, so this is a
// pure streaming pass; it moves the f32->bf16 conversion out of the
// compute-bound GEMM into a bandwidth-bound pass (~1.5us at 23.3 TB/s).
// ---------------------------------------------------------------------------
__global__ void mha_pack_x(const float* __restrict__ X, uint32_t* __restrict__ Xp) {
    int idx = blockIdx.x * blockDim.x + threadIdx.x;     // over MROWS*(D_MODEL/2)
    if (idx >= MROWS * (D_MODEL / 2)) return;
    float lo = X[2 * idx];
    float hi = X[2 * idx + 1];
    Xp[idx] = pack_bf16_rne(lo, hi);
}

// ---------------------------------------------------------------------------
// QKV projection: Xp bf16-packed [8192,256 dwords] @ Wp(bf16 packed) + bias
// -> bf16 head-split out[((b*8 + h)*2048 + s)*64 + d].
// One wave computes a 32(M) x 64(N) tile: each B fragment feeds two WMMAs,
// and the inner loop is pure b32 loads + WMMA (no conversion VALU).
// ---------------------------------------------------------------------------
__global__ __launch_bounds__(128) void mha_proj_qkv(
        const uint32_t* __restrict__ Xp, const uint32_t* __restrict__ Wp,
        const float* __restrict__ bias, uint16_t* __restrict__ outh) {
    const int lane = threadIdx.x & 31;
    const int wv   = threadIdx.x >> 5;
    const int tile = blockIdx.x * 4 + wv;       // 2048 tiles total
    const int tm = tile >> 3, tn = tile & 7;    // 256 row-tiles x 8 col-tiles
    const int row0 = tm * 32, col0 = tn * 64;
    const int half = lane >> 4, l15 = lane & 15;

    f32x8 acc[2][4] = {};
    const uint32_t* ar0 = Xp + (row0 + l15) * (D_MODEL / 2);
    const uint32_t* ar1 = ar0 + 16 * (D_MODEL / 2);
    for (int kb = 0; kb < D_MODEL; kb += 32) {
        FragB a0, a1;
        #pragma unroll
        for (int i = 0; i < 8; ++i) {   // A: 16-bit 16x32 layout
            int k0 = kb + (half << 4) + (i << 1);
            a0.u[i] = ar0[k0 >> 1];
            a1.u[i] = ar1[k0 >> 1];
        }
        #pragma unroll
        for (int nt = 0; nt < 4; ++nt) {
            FragB b;
            int col = col0 + nt * 16 + l15;
            #pragma unroll
            for (int i = 0; i < 8; ++i) {   // B: lanes0-15 K=0..15, lanes16-31 K=16..31
                int k0 = kb + (half << 4) + (i << 1);
                b.u[i] = Wp[(k0 >> 1) * D_MODEL + col];
            }
            acc[0][nt] = wmma_bf16(a0, b, acc[0][nt]);
            acc[1][nt] = wmma_bf16(a1, b, acc[1][nt]);
        }
    }
    #pragma unroll
    for (int mh = 0; mh < 2; ++mh)
        #pragma unroll
        for (int nt = 0; nt < 4; ++nt) {
            int col = col0 + nt * 16 + l15;
            float bs = bias[col];
            int h = col >> 6, d = col & 63;
            #pragma unroll
            for (int i = 0; i < 8; ++i) {
                int row = row0 + mh * 16 + i + (half << 3);
                int bb = row >> 11, s = row & (S_LEN - 1);
                outh[(((bb * NH + h) * S_LEN) + s) * DKH + d] =
                    bf16u(acc[mh][nt][i] + bs);
            }
        }
}

// ---------------------------------------------------------------------------
// Flash attention: one wave handles 16 query rows of one (b,h); keys in blocks
// of 32. K/V tiles (4KB each) are double-buffered through LDS with
// GLOBAL_LOAD_ASYNC_TO_LDS_B128 so block n+1 streams in behind block n's
// WMMAs; gated with s_wait_asynccnt. Writes ctx bf16 [B*S, 512].
// ---------------------------------------------------------------------------
__global__ __launch_bounds__(32) void mha_attn(
        const uint32_t* __restrict__ Qh, const uint16_t* __restrict__ Khu,
        const uint16_t* __restrict__ Vh, uint16_t* __restrict__ Ctx) {
    __shared__ __align__(16) uint16_t kt[2][32 * DKH];   // 2 x 4KB
    __shared__ __align__(16) uint16_t vt[2][32 * DKH];   // 2 x 4KB
    __shared__ __align__(16) uint16_t pl[16 * 32];       // P staging, 1KB
    const int bh = blockIdx.y;
    const int q0 = blockIdx.x * 16;
    const int lane = threadIdx.x, half = lane >> 4, l15 = lane & 15;

    const uint32_t* Q  = Qh  + bh * (S_LEN * (DKH / 2));
    const uint16_t* Kb = Khu + bh * (S_LEN * DKH);
    const uint16_t* Vb = Vh  + bh * (S_LEN * DKH);

    FragB aq[2];
    #pragma unroll
    for (int kc = 0; kc < 2; ++kc)
        #pragma unroll
        for (int i = 0; i < 8; ++i) {
            int d0 = kc * 32 + (half << 4) + (i << 1);
            aq[kc].u[i] = Q[(q0 + l15) * (DKH / 2) + (d0 >> 1)];
        }

    f32x8 acc[4] = {};
    float runm[8], runs[8];
    #pragma unroll
    for (int i = 0; i < 8; ++i) { runm[i] = -__builtin_inff(); runs[i] = 0.f; }

    // prime buffer 0
    async_copy_4k(Kb, kt[0], lane);
    async_copy_4k(Vb, vt[0], lane);

    int buf = 0;
    for (int kb = 0; kb < S_LEN; kb += 32, buf ^= 1) {
        if (kb + 32 < S_LEN) {
            // stream next block into the other buffer, then wait only for ours
            async_copy_4k(Kb + (kb + 32) * DKH, kt[buf ^ 1], lane);
            async_copy_4k(Vb + (kb + 32) * DKH, vt[buf ^ 1], lane);
            wait_async_le16();
        } else {
            wait_async_0();
        }
        const uint32_t* ku = (const uint32_t*)kt[buf];
        const uint16_t* vu = vt[buf];

        f32x8 sc[2];
        #pragma unroll
        for (int ns = 0; ns < 2; ++ns) {
            f32x8 s = {};
            #pragma unroll
            for (int kc = 0; kc < 2; ++kc) {
                FragB bk;   // B = K^T chunk (32 x 16) read from LDS
                #pragma unroll
                for (int i = 0; i < 8; ++i) {
                    int k0 = kc * 32 + (half << 4) + (i << 1);
                    bk.u[i] = ku[(ns * 16 + l15) * (DKH / 2) + (k0 >> 1)];
                }
                s = wmma_bf16(aq[kc], bk, s);
            }
            sc[ns] = s;
        }
        // online softmax over the 16x32 score tile
        #pragma unroll
        for (int i = 0; i < 8; ++i) {
            float s0 = sc[0][i] * 0.125f;      // 1/sqrt(64)
            float s1 = sc[1][i] * 0.125f;
            float tmax = rmax16(fmaxf(s0, s1));
            float newm = fmaxf(runm[i], tmax);
            float corr = __expf(runm[i] - newm);
            float p0 = __expf(s0 - newm);
            float p1 = __expf(s1 - newm);
            float ts = rsum16(p0 + p1);
            runs[i] = runs[i] * corr + ts;
            runm[i] = newm;
            acc[0][i] *= corr; acc[1][i] *= corr;
            acc[2][i] *= corr; acc[3][i] *= corr;
            int m = i + (half << 3);
            pl[m * 32 + l15]      = bf16u(p0);
            pl[m * 32 + 16 + l15] = bf16u(p1);
        }
        // re-read P in A-fragment layout (DS ops are in-order within a wave)
        FragB pa;
        const uint32_t* plu = (const uint32_t*)pl;
        #pragma unroll
        for (int i = 0; i < 8; ++i) {
            int k0 = (half << 4) + (i << 1);
            pa.u[i] = plu[l15 * 16 + (k0 >> 1)];
        }
        // ctx += P (16x32) @ V_block (32 x 64), V read from LDS
        #pragma unroll
        for (int nt = 0; nt < 4; ++nt) {
            FragB bv;
            #pragma unroll
            for (int i = 0; i < 8; ++i) {
                int k0 = (half << 4) + (i << 1);
                int d  = nt * 16 + l15;
                uint32_t lo = vu[k0 * DKH + d];
                uint32_t hi = vu[(k0 + 1) * DKH + d];
                bv.u[i] = lo | (hi << 16);
            }
            acc[nt] = wmma_bf16(pa, bv, acc[nt]);
        }
    }
    // epilogue: normalize and store ctx bf16 [B*S, D_MODEL]
    const int b = bh >> 3, h = bh & 7;
    #pragma unroll
    for (int nt = 0; nt < 4; ++nt)
        #pragma unroll
        for (int i = 0; i < 8; ++i) {
            int m = i + (half << 3);
            int srow = q0 + m;
            float v = acc[nt][i] / runs[i];
            Ctx[(b * S_LEN + srow) * D_MODEL + h * DKH + nt * 16 + l15] = bf16u(v);
        }
}

// ---------------------------------------------------------------------------
// Output projection: ctx bf16 [8192,512] @ Wo(bf16 packed) + bo -> fp32 out
// 32(M) x 64(N) tile per wave, same B-fragment reuse as mha_proj_qkv.
// ---------------------------------------------------------------------------
__global__ __launch_bounds__(128) void mha_proj_out(
        const uint32_t* __restrict__ Ctx, const uint32_t* __restrict__ Wp,
        const float* __restrict__ bias, float* __restrict__ out) {
    const int lane = threadIdx.x & 31;
    const int wv   = threadIdx.x >> 5;
    const int tile = blockIdx.x * 4 + wv;
    const int tm = tile >> 3, tn = tile & 7;
    const int row0 = tm * 32, col0 = tn * 64;
    const int half = lane >> 4, l15 = lane & 15;

    f32x8 acc[2][4] = {};
    const uint32_t* ar0 = Ctx + (row0 + l15) * (D_MODEL / 2);
    const uint32_t* ar1 = ar0 + 16 * (D_MODEL / 2);
    for (int kb = 0; kb < D_MODEL; kb += 32) {
        FragB a0, a1;
        #pragma unroll
        for (int i = 0; i < 8; ++i) {
            int k0 = kb + (half << 4) + (i << 1);
            a0.u[i] = ar0[k0 >> 1];
            a1.u[i] = ar1[k0 >> 1];
        }
        #pragma unroll
        for (int nt = 0; nt < 4; ++nt) {
            FragB b;
            int col = col0 + nt * 16 + l15;
            #pragma unroll
            for (int i = 0; i < 8; ++i) {
                int k0 = kb + (half << 4) + (i << 1);
                b.u[i] = Wp[(k0 >> 1) * D_MODEL + col];
            }
            acc[0][nt] = wmma_bf16(a0, b, acc[0][nt]);
            acc[1][nt] = wmma_bf16(a1, b, acc[1][nt]);
        }
    }
    #pragma unroll
    for (int mh = 0; mh < 2; ++mh)
        #pragma unroll
        for (int nt = 0; nt < 4; ++nt) {
            int col = col0 + nt * 16 + l15;
            float bs = bias[col];
            #pragma unroll
            for (int i = 0; i < 8; ++i) {
                int row = row0 + mh * 16 + i + (half << 3);
                out[row * D_MODEL + col] = acc[mh][nt][i] + bs;
            }
        }
}

extern "C" void kernel_launch(void* const* d_in, const int* in_sizes, int n_in,
                              void* d_out, int out_size, void* d_ws, size_t ws_size,
                              hipStream_t stream) {
    const float* q_in = (const float*)d_in[0];
    const float* k_in = (const float*)d_in[1];
    const float* v_in = (const float*)d_in[2];
    const float* Wq = (const float*)d_in[3];
    const float* bq = (const float*)d_in[4];
    const float* Wk = (const float*)d_in[5];
    const float* bk = (const float*)d_in[6];
    const float* Wv = (const float*)d_in[7];
    const float* bv = (const float*)d_in[8];
    const float* Wo = (const float*)d_in[9];
    const float* bo = (const float*)d_in[10];

    char* ws = (char*)d_ws;
    // bf16 tensors: B*H*S*64 = 8192*512 elements = 8 MB each
    const size_t SZ_T = (size_t)MROWS * D_MODEL * sizeof(uint16_t);          // 8 MB
    const size_t SZ_W = (size_t)(D_MODEL / 2) * D_MODEL * sizeof(uint32_t);  // 512 KB

    uint16_t* qh  = (uint16_t*)(ws);
    uint16_t* kh  = (uint16_t*)(ws + SZ_T);
    uint16_t* vh  = (uint16_t*)(ws + 2 * SZ_T);
    uint16_t* ctx = (uint16_t*)(ws + 3 * SZ_T);
    uint32_t* wqp = (uint32_t*)(ws + 4 * SZ_T);
    uint32_t* wkp = (uint32_t*)(ws + 4 * SZ_T + SZ_W);
    uint32_t* wvp = (uint32_t*)(ws + 4 * SZ_T + 2 * SZ_W);
    uint32_t* wop = (uint32_t*)(ws + 4 * SZ_T + 3 * SZ_W);
    uint32_t* xp  = (uint32_t*)(ws + 4 * SZ_T + 4 * SZ_W);   // reused for q/k/v

    // 1) repack weights to bf16 pair-interleaved
    const int npack = (D_MODEL / 2) * D_MODEL;
    const int pb = (npack + 255) / 256;
    mha_pack_w<<<pb, 256, 0, stream>>>(Wq, wqp);
    mha_pack_w<<<pb, 256, 0, stream>>>(Wk, wkp);
    mha_pack_w<<<pb, 256, 0, stream>>>(Wv, wvp);
    mha_pack_w<<<pb, 256, 0, stream>>>(Wo, wop);

    // 2) per input: stream-convert activations to packed bf16, then project.
    //    xp is reused across q/k/v (stream-ordered).
    const int nxp = MROWS * (D_MODEL / 2);
    const int xb = (nxp + 255) / 256;
    const int proj_blocks = (MROWS / 32) * (D_MODEL / 64) / 4;   // 512
    mha_pack_x<<<xb, 256, 0, stream>>>(q_in, xp);
    mha_proj_qkv<<<proj_blocks, 128, 0, stream>>>(xp, wqp, bq, qh);
    mha_pack_x<<<xb, 256, 0, stream>>>(k_in, xp);
    mha_proj_qkv<<<proj_blocks, 128, 0, stream>>>(xp, wkp, bk, kh);
    mha_pack_x<<<xb, 256, 0, stream>>>(v_in, xp);
    mha_proj_qkv<<<proj_blocks, 128, 0, stream>>>(xp, wvp, bv, vh);

    // 3) flash attention (async double-buffered K/V through LDS)
    dim3 agrid(S_LEN / 16, NB * NH);
    mha_attn<<<agrid, 32, 0, stream>>>((const uint32_t*)qh, kh, vh, ctx);

    // 4) output projection -> fp32 d_out
    mha_proj_out<<<proj_blocks, 128, 0, stream>>>((const uint32_t*)ctx, wop, bo,
                                                  (float*)d_out);
}
